// ScaledDotProductAttention_24412594111141
// MI455X (gfx1250) — matrix operand order repaired and checked by reading, not compiled, not run on hardware
//
#include <hip/hip_runtime.h>

#define BB 2
#define HH 16
#define LL 2048
#define DD 64
#define BLOCK_M 64
#define BLOCK_N 64
#define NWAVES 4
#define NKB (LL / BLOCK_N)
#define VSTRIDE (BLOCK_N + 8)   // 72 elements = 144B rows (16B aligned, conflict-reducing)

typedef __bf16 bf16_t;
typedef bf16_t v16bf __attribute__((ext_vector_type(16)));
typedef bf16_t v8bf  __attribute__((ext_vector_type(8)));
typedef bf16_t v4bf  __attribute__((ext_vector_type(4)));
typedef bf16_t v2bf  __attribute__((ext_vector_type(2)));
typedef float  v8f   __attribute__((ext_vector_type(8)));
typedef float  v4f   __attribute__((ext_vector_type(4)));
typedef int    v4i   __attribute__((ext_vector_type(4)));

#define L2E   1.4426950408889634f       // log2(e)
#define QSC   (0.125f * L2E)            // 1/sqrt(64) * log2(e)
#define MNEG  (-10000.0f * L2E)         // masked score in exp2 domain

// single v_exp_f32
static __device__ __forceinline__ float exp2_hw(float x) {
  return __builtin_amdgcn_exp2f(x);
}

static __device__ __forceinline__ bf16_t f2bf(float f) {
  union { float f; unsigned u; } v; v.f = f;
  unsigned r = (v.u + 0x7FFFu + ((v.u >> 16) & 1u)) >> 16;
  unsigned short h = (unsigned short)r;
  bf16_t b;
  __builtin_memcpy(&b, &h, 2);
  return b;
}

// Packed f32x2 -> bf16x2; prefer the HW packed convert if the backend has it.
static __device__ __forceinline__ v2bf pk2(float a, float b) {
#if __has_builtin(__builtin_amdgcn_cvt_pk_bf16_f32)
  return __builtin_amdgcn_cvt_pk_bf16_f32(a, b);
#else
  v2bf r; r[0] = f2bf(a); r[1] = f2bf(b); return r;
#endif
}

// 8 floats -> v8bf via packed converts + shufflevector tree (no b16 move chains)
static __device__ __forceinline__ v8bf pk8(const float* x) {
  v2bf p0 = pk2(x[0], x[1]), p1 = pk2(x[2], x[3]);
  v2bf p2 = pk2(x[4], x[5]), p3 = pk2(x[6], x[7]);
  v4bf q0 = __builtin_shufflevector(p0, p1, 0, 1, 2, 3);
  v4bf q1 = __builtin_shufflevector(p2, p3, 0, 1, 2, 3);
  return __builtin_shufflevector(q0, q1, 0, 1, 2, 3, 4, 5, 6, 7);
}

static __device__ __forceinline__ v16bf cmb16(v8bf a, v8bf b) {
  return __builtin_shufflevector(a, b, 0, 1, 2, 3, 4, 5, 6, 7,
                                 8, 9, 10, 11, 12, 13, 14, 15);
}

// A-operand fragment: halves 0..7 at p[0..7], halves 8..15 at p[8..15]
static __device__ __forceinline__ v16bf ldA(const bf16_t* p) {
  v8bf a = *(const v8bf*)p;
  v8bf b = *(const v8bf*)(p + 8);
  return cmb16(a, b);
}

__global__ __launch_bounds__(128)
void fa_fwd_bf16wmma(const float* __restrict__ Q, const float* __restrict__ K,
                     const float* __restrict__ V, const float* __restrict__ bias,
                     const int* __restrict__ mask, float* __restrict__ out) {
  // Shared staging: K block row-major [key][d], V block transposed [d][key]
  __shared__ __attribute__((aligned(16))) bf16_t klds[BLOCK_N * DD];
  __shared__ __attribute__((aligned(16))) bf16_t vlds[DD * VSTRIDE];

  const int tid    = threadIdx.x;
  const int lane   = tid & 31;
  const int wave   = tid >> 5;
  const int half16 = lane >> 4;    // 0: lanes 0-15, 1: lanes 16-31
  const int ln     = lane & 15;

  const int bh = blockIdx.y;       // 0..31
  const int b  = bh / HH;
  const int h  = bh % HH;
  const int q0 = blockIdx.x * BLOCK_M + wave * 16;
  const int q  = q0 + ln;          // this lane's query column (Sᵀ formulation)

  const float* Qbase = Q    + (size_t)bh * LL * DD;
  const float* Kbase = K    + (size_t)bh * LL * DD;
  const float* Vbase = V    + (size_t)bh * LL * DD;
  const float* Bbase = bias + (size_t)h  * LL * LL;
  const int*   Mbase = mask + (size_t)b  * LL;

  // ---- Qᵀ B-fragments (32 d x 16 q), scale*log2e folded in.
  // B frag k order used here (contiguous, same as A): lane col n = q (ln); lanes0-15 K=c*32+0..15,
  // lanes16-31 K=c*32+16..31; halves t = K offset t.
  v16bf qf[2];
  {
    const float* qrow = Qbase + (size_t)q * DD + half16 * 16;
    #pragma unroll
    for (int c = 0; c < 2; ++c) {
      float tq[16];
      #pragma unroll
      for (int t = 0; t < 16; ++t) tq[t] = qrow[c * 32 + t] * QSC;
      qf[c] = cmb16(pk8(&tq[0]), pk8(&tq[8]));
    }
  }

  // Online softmax state (exp2 domain): one query column per lane.
  float m_s = -3.0e38f, l_s = 0.0f;
  v8f o[4];
  #pragma unroll
  for (int j = 0; j < 4; ++j) { v8f z = {}; o[j] = z; }

  for (int kb = 0; kb < NKB; ++kb) {
    const int k0 = kb * BLOCK_N;

    // ---------- cooperative staging: K -> klds[key][d], V -> vlds[d][key] (bf16)
    #pragma unroll
    for (int t = 0; t < 8; ++t) {
      const int f   = t * 128 + tid;   // float4 index within 64x64 block
      const int key = f >> 4;          // 0..63
      const int d4  = (f & 15) << 2;   // 0,4,...,60

      const float4 kq = *(const float4*)(Kbase + (size_t)(k0 + key) * DD + d4);
      v2bf kp0 = pk2(kq.x, kq.y), kp1 = pk2(kq.z, kq.w);
      *(v4bf*)(&klds[key * DD + d4]) =
          __builtin_shufflevector(kp0, kp1, 0, 1, 2, 3);

      const float4 vq = *(const float4*)(Vbase + (size_t)(k0 + key) * DD + d4);
      v2bf vp0 = pk2(vq.x, vq.y), vp1 = pk2(vq.z, vq.w);
      vlds[(d4 + 0) * VSTRIDE + key] = vp0[0];
      vlds[(d4 + 1) * VSTRIDE + key] = vp0[1];
      vlds[(d4 + 2) * VSTRIDE + key] = vp1[0];
      vlds[(d4 + 3) * VSTRIDE + key] = vp1[1];
    }

    // ---------- bias/mask for this lane's q column: contiguous along keys
    // key = k0 + tt*16 + half16*8 + r  -> two float4 runs per tile
    v4f bv[4][2];
    v4i mv[4][2];
    {
      const float* bp = Bbase + (size_t)q * LL + k0 + half16 * 8;
      const int*   mp = Mbase + k0 + half16 * 8;
      #pragma unroll
      for (int tt = 0; tt < 4; ++tt) {
        bv[tt][0] = *(const v4f*)(bp + tt * 16);
        bv[tt][1] = *(const v4f*)(bp + tt * 16 + 4);
        mv[tt][0] = *(const v4i*)(mp + tt * 16);
        mv[tt][1] = *(const v4i*)(mp + tt * 16 + 4);
      }
    }

    // keep next block's bias lines streaming from HBM (dominant traffic)
    if (kb + 1 < NKB) {
      const float* pf = Bbase + (size_t)q * LL + (k0 + 64);
      __builtin_prefetch(pf, 0, 0);
      __builtin_prefetch(pf + 32, 0, 0);
    }

    __syncthreads();

    // ---------- Sᵀ = K * Qᵀ : 4 key tiles of 16 rows, each = 2 chained k=32 WMMAs
    // A-fragment = K tile (16 keys x 32 d) from klds rows.
    v8f s[4];
    #pragma unroll
    for (int tt = 0; tt < 4; ++tt) {
      const bf16_t* kr = &klds[(tt * 16 + ln) * DD + (half16 ? 16 : 0)];
      v16bf a0 = ldA(kr);        // d chunk 0..31
      v16bf a1 = ldA(kr + 32);   // d chunk 32..63
      v8f acc = {};
      acc = __builtin_amdgcn_wmma_f32_16x16x32_bf16(false, a0, false, qf[0],
                                                    (short)0, acc, false, false);
      acc = __builtin_amdgcn_wmma_f32_16x16x32_bf16(false, a1, false, qf[1],
                                                    (short)0, acc, false, false);
      s[tt] = acc;   // reg r: key = k0 + tt*16 + half16*8 + r, col q (x log2e)
    }

    // ---------- mask (key-wise, replace) then bias (fma by log2e) -- exp2 domain
    #pragma unroll
    for (int tt = 0; tt < 4; ++tt) {
      #pragma unroll
      for (int r = 0; r < 8; ++r) {
        float sc = s[tt][r];
        if (mv[tt][r >> 2][r & 3] == 0) sc = MNEG;
        sc = __builtin_fmaf(bv[tt][r >> 2][r & 3], L2E, sc);
        s[tt][r] = sc;
      }
    }

    // ---------- online softmax: column is lane-local + one cross-half shuffle
    float mx = s[0][0];
    #pragma unroll
    for (int tt = 0; tt < 4; ++tt)
      #pragma unroll
      for (int r = 0; r < 8; ++r) mx = fmaxf(mx, s[tt][r]);
    mx = fmaxf(mx, __shfl_xor(mx, 16, 32));

    const float mnew = fmaxf(m_s, mx);
    const float a = exp2_hw(m_s - mnew);
    float rs = 0.0f;
    #pragma unroll
    for (int tt = 0; tt < 4; ++tt) {
      #pragma unroll
      for (int r = 0; r < 8; ++r) {
        float p = exp2_hw(s[tt][r] - mnew);
        s[tt][r] = p;
        rs += p;
      }
    }
    rs += __shfl_xor(rs, 16, 32);
    l_s = l_s * a + rs;
    m_s = mnew;

    #pragma unroll
    for (int j = 0; j < 4; ++j)
      #pragma unroll
      for (int r = 0; r < 8; ++r) o[j][r] *= a;

    // ---------- Pᵀ B-fragments straight from registers (half-swap, no LDS)
    // B-frag c needs keys c*32+0..15 (lanes<16) / c*32+16..31 (lanes>=16):
    //   lanes<16 : low = own s[2c][0..7],            high = partner s[2c][0..7]
    //   lanes>=16: low = partner s[2c+1][0..7],      high = own s[2c+1][0..7]
    v16bf pfr[2];
    #pragma unroll
    for (int c = 0; c < 2; ++c) {
      float low[8], high[8];
      #pragma unroll
      for (int r = 0; r < 8; ++r) {
        float g0 = __shfl_xor(s[2 * c][r], 16, 32);
        float g1 = __shfl_xor(s[2 * c + 1][r], 16, 32);
        low[r]  = half16 ? g1 : s[2 * c][r];
        high[r] = half16 ? s[2 * c + 1][r] : g0;
      }
      pfr[c] = cmb16(pk8(low), pk8(high));
    }

    // ---------- Oᵀ += Vᵀ * Pᵀ : 4 d-tiles, each = 2 chained k=32 WMMAs
    // A-fragment = Vᵀ tile (16 d x 32 keys) from vlds rows.
    #pragma unroll
    for (int j = 0; j < 4; ++j) {
      const bf16_t* vr = &vlds[(j * 16 + ln) * VSTRIDE + (half16 ? 16 : 0)];
      v16bf a0 = ldA(vr);        // keys k0 .. k0+31
      v16bf a1 = ldA(vr + 32);   // keys k0+32 .. k0+63
      o[j] = __builtin_amdgcn_wmma_f32_16x16x32_bf16(false, a0, false, pfr[0],
                                                     (short)0, o[j], false, false);
      o[j] = __builtin_amdgcn_wmma_f32_16x16x32_bf16(false, a1, false, pfr[1],
                                                     (short)0, o[j], false, false);
    }

    __syncthreads();  // all waves done with klds/vlds before next staging pass
  }

  // ---------- epilogue: lane owns row q of O; contiguous b128 stores
  const float inv = 1.0f / l_s;
  float* orow = out + ((size_t)bh * LL + q) * DD + half16 * 8;
  #pragma unroll
  for (int j = 0; j < 4; ++j) {
    v4f w0 = {o[j][0] * inv, o[j][1] * inv, o[j][2] * inv, o[j][3] * inv};
    v4f w1 = {o[j][4] * inv, o[j][5] * inv, o[j][6] * inv, o[j][7] * inv};
    *(v4f*)(orow + j * 16)     = w0;
    *(v4f*)(orow + j * 16 + 4) = w1;
  }
}

extern "C" void kernel_launch(void* const* d_in, const int* in_sizes, int n_in,
                              void* d_out, int out_size, void* d_ws, size_t ws_size,
                              hipStream_t stream) {
  const float* Q    = (const float*)d_in[0];
  const float* K    = (const float*)d_in[1];
  const float* V    = (const float*)d_in[2];
  const float* bias = (const float*)d_in[3];
  const int*   mask = (const int*)d_in[4];
  float* out = (float*)d_out;

  dim3 grid(LL / BLOCK_M, BB * HH);
  fa_fwd_bf16wmma<<<grid, dim3(128), 0, stream>>>(Q, K, V, bias, mask, out);

  (void)in_sizes; (void)n_in; (void)out_size; (void)d_ws; (void)ws_size;
}
